// SparseTemporalMemory_41970420417620
// MI455X (gfx1250) — matrix operand order repaired and checked by reading, not compiled, not run on hardware
//
#include <hip/hip_runtime.h>
#include <math.h>

// ---------------- constants from the reference ----------------
namespace {
constexpr int B_  = 32;
constexpr int M_  = 16384;
constexpr int WC  = 256;
constexpr int R_  = 8;
constexpr int K_  = 16;
constexpr int KL_ = 16;
constexpr int IN_ = 1024;
constexpr int TKL = 2 * KL_;            // 32
constexpr int C_  = R_ * K_ + TKL + 1;  // 161
constexpr int NOUT = R_ * WC + WC + C_ + 1; // 2466 fused GEMM rows
}

typedef __attribute__((ext_vector_type(16))) _Float16 v16h;
typedef __attribute__((ext_vector_type(8)))  _Float16 v8h;
typedef __attribute__((ext_vector_type(8)))  float    v8f;

// ============================================================
// K1: fused small GEMMs:  out_row = W[row] . x[b] + bias  for all b
//   rows [0,2048)  -> read_query        (raw)
//   rows [2048,2304)-> write_vector     (raw)
//   rows [2304,2465)-> gate             (sigmoid)
//   row  2465       -> write_gate       (sigmoid)
// ============================================================
__global__ void k1_gemms(const float* __restrict__ x,
                         const float* __restrict__ W_rq, const float* __restrict__ b_rq,
                         const float* __restrict__ W_wv, const float* __restrict__ b_wv,
                         const float* __restrict__ W_ig, const float* __restrict__ b_ig,
                         const float* __restrict__ W_wg, const float* __restrict__ b_wg,
                         float* __restrict__ rqbuf, float* __restrict__ wvbuf,
                         float* __restrict__ gatebuf, float* __restrict__ wgbuf) {
  __shared__ float xs[B_ * 256];           // 32KB: all batches, one 256-wide k-chunk
  const int t   = threadIdx.x;
  const int row = blockIdx.x * 256 + t;

  const float* Wp = nullptr; const float* bp = nullptr;
  int lr = 0, kind = -1;
  if (row < NOUT) {
    if      (row < 2048) { Wp = W_rq; bp = b_rq; lr = row;        kind = 0; }
    else if (row < 2304) { Wp = W_wv; bp = b_wv; lr = row - 2048; kind = 1; }
    else if (row < 2465) { Wp = W_ig; bp = b_ig; lr = row - 2304; kind = 2; }
    else                 { Wp = W_wg; bp = b_wg; lr = 0;          kind = 3; }
  }

  float acc[B_];
  #pragma unroll
  for (int b = 0; b < B_; ++b) acc[b] = 0.0f;

  for (int kc = 0; kc < IN_ / 256; ++kc) {
    __syncthreads();
    // cooperative stage of x[:, kc*256 : kc*256+256]
    for (int i = 0; i < B_; ++i) {
      int f = i * 256 + t;
      int b = f >> 8, k = f & 255;
      xs[f] = x[(size_t)b * IN_ + kc * 256 + k];
    }
    __syncthreads();
    if (kind >= 0) {
      const float4* w4  = (const float4*)(Wp + (size_t)lr * IN_ + kc * 256);
      const float4* xs4 = (const float4*)xs;
      for (int kk = 0; kk < 64; ++kk) {
        float4 w = w4[kk];
        #pragma unroll
        for (int b = 0; b < B_; ++b) {
          float4 xv = xs4[b * 64 + kk];
          acc[b] += w.x * xv.x + w.y * xv.y + w.z * xv.z + w.w * xv.w;
        }
      }
    }
  }

  if (kind >= 0) {
    float bias = bp[lr];
    #pragma unroll
    for (int b = 0; b < B_; ++b) {
      float o = acc[b] + bias;
      if (kind >= 2) o = 1.0f / (1.0f + __expf(-o));
      if      (kind == 0) rqbuf [b * (R_ * WC) + row]        = o;
      else if (kind == 1) wvbuf [b * WC + lr]                = o;
      else if (kind == 2) gatebuf[b * C_ + lr]               = o;
      else                wgbuf [b]                          = o;
    }
  }
}

// ============================================================
// K2: per-batch state update (everything that feeds the output)
// ============================================================
__device__ __forceinline__ void top16_virtual(const float* vals /*32*/, int* out /*16*/) {
  // jax.lax.top_k over [vals[0..31], 0, 0, ...]: (val desc, idx asc)
  bool used[TKL];
  for (int i = 0; i < TKL; ++i) used[i] = false;
  int nextvirt = TKL;
  for (int s = 0; s < KL_; ++s) {
    float bv = -3.4e38f; int bi = 0x7fffffff;
    for (int m = 0; m < TKL; ++m)
      if (!used[m] && (vals[m] > bv || (vals[m] == bv && m < bi))) { bv = vals[m]; bi = m; }
    if (bi != 0x7fffffff && (bv > 0.0f || (bv == 0.0f && bi < nextvirt))) {
      used[bi] = true; out[s] = bi;
    } else {
      out[s] = nextvirt++;        // virtual zero entry at idx >= 32
    }
  }
}

__global__ void k2_state(const float* __restrict__ vism,       // visible_memory
                         const float* __restrict__ link,
                         const float* __restrict__ rev,
                         const float* __restrict__ prec,
                         const float* __restrict__ rws,
                         const float* __restrict__ wws,
                         const float* __restrict__ usage,
                         const int*   __restrict__ read_positions,
                         const int*   __restrict__ least_used,
                         const float* __restrict__ gatebuf,
                         const float* __restrict__ wgbuf,
                         const float* __restrict__ wvbuf,
                         float* __restrict__ visnew,
                         int*   __restrict__ rowmap,
                         int*   __restrict__ posbuf) {
  __shared__ int   s_rp[C_];
  __shared__ float s_rwvis[C_], s_relu[C_], s_imin[C_], s_wwnew[C_];
  __shared__ int   s_owner[C_];
  __shared__ float s_red[256];
  __shared__ float s_fv[TKL], s_bv[TKL];

  const int b = blockIdx.x, t = threadIdx.x;

  if (t < C_) {
    int rp = read_positions[b * C_ + t];
    s_rp[t]    = rp;
    s_rwvis[t] = rws[(size_t)b * M_ + rp] + 1.0f;
    s_relu[t]  = usage[(size_t)b * M_ + rp];
  }
  __syncthreads();

  s_red[t] = (t < C_) ? s_relu[t] : 3.0e38f;
  __syncthreads();
  for (int off = 128; off > 0; off >>= 1) {
    if (t < off) s_red[t] = fminf(s_red[t], s_red[t + off]);
    __syncthreads();
  }
  const float umin = s_red[0];
  __syncthreads();

  if (t < C_) {
    float im = (s_relu[t] == umin) ? 1.0f : 0.0f;
    s_imin[t] = im;
    float g = gatebuf[b * C_ + t];
    s_wwnew[t] = wgbuf[b] * (g * s_rwvis[t] + (1.0f - g) * im);
    int own = 1;                               // last-occurrence wins scatter
    for (int c2 = t + 1; c2 < C_; ++c2)
      if (s_rp[c2] == s_rp[t]) { own = 0; break; }
    s_owner[t] = own;
  }
  __syncthreads();

  // visible_memory update -> vis_new
  for (int idx = t; idx < C_ * WC; idx += 256) {
    int c = idx >> 8, w = idx & 255;
    float v = vism[(size_t)b * C_ * WC + idx] * (1.0f - s_imin[c])
            + s_wwnew[c] * wvbuf[b * WC + w];
    visnew[(size_t)b * C_ * WC + idx] = v;
  }
  // rowmap: -1 everywhere, owner c at scattered rows
  for (int m = t; m < M_; m += 256) rowmap[(size_t)b * M_ + m] = -1;
  __syncthreads();
  if (t < C_ && s_owner[t]) rowmap[(size_t)b * M_ + s_rp[t]] = t;

  // fwd/bwd values: only diagonal of first 32 rows survives eye(M,TKL)
  if (t < TKL) {
    const int m   = t;
    const int trp = s_rp[R_ * K_ + 1 + m];
    const float trw = rws[(size_t)b * M_ + trp];

    float www = wws[(size_t)b * M_ + m];                   // updated wws at m
    for (int c = 0; c < C_; ++c)
      if (s_owner[c] && s_rp[c] == m) www = s_wwnew[c];
    float ld = link[((size_t)b * M_ + m) * TKL + m];
    s_fv[m] = ((1.0f - www) * ld + www * prec[b * TKL + m]) * trw;

    float tw = wws[(size_t)b * M_ + trp];                  // updated wws at trp
    for (int c = 0; c < C_; ++c)
      if (s_owner[c] && s_rp[c] == trp) tw = s_wwnew[c];
    float pd = 0.0f;                                       // prec_dense at m
    for (int j = TKL - 1; j >= 0; --j)
      if (s_rp[R_ * K_ + 1 + j] == m) { pd = prec[b * TKL + j]; break; }
    float rd = rev[((size_t)b * M_ + m) * TKL + m];
    s_bv[m] = ((1.0f - tw) * rd + tw * pd) * trw;
  }
  __syncthreads();

  if (t == 0) { int o[KL_]; top16_virtual(s_fv, o);
                for (int i = 0; i < KL_; ++i) posbuf[b * C_ + R_ * K_ + i] = o[i]; }
  if (t == 1) { int o[KL_]; top16_virtual(s_bv, o);
                for (int i = 0; i < KL_; ++i) posbuf[b * C_ + R_ * K_ + KL_ + i] = o[i]; }
  if (t == 2)   posbuf[b * C_ + C_ - 1] = least_used[b];
}

// ============================================================
// K3: scores[b,r,m] = <rq[b,r,:], memory[b,m,:]>  via WMMA f16->f32
//   per wave: 16 memory rows (A), rq f16 in LDS zero-padded to 16 rows (B)
// ============================================================
__global__ void k3_scores_wmma(const float* __restrict__ memory,
                               const float* __restrict__ rqbuf,
                               float* __restrict__ scores) {
  __shared__ __align__(32) _Float16 rqh[16 * WC];   // 8KB, rows 8..15 zero
  const int b = blockIdx.y, t = threadIdx.x;

  for (int i = t; i < 16 * WC; i += 256) {
    int r = i >> 8, w = i & 255;
    rqh[i] = (r < R_) ? (_Float16)rqbuf[b * (R_ * WC) + r * WC + w] : (_Float16)0.0f;
  }
  __syncthreads();

  const int wid = t >> 5, lane = t & 31;
  const int half = lane >> 4, lrow = lane & 15;
  const int rowbase = blockIdx.x * 128 + wid * 16;
  const float* arow = memory + ((size_t)b * M_ + rowbase + lrow) * WC;

  v8f acc = {};
  #pragma unroll
  for (int chunk = 0; chunk < 8; ++chunk) {
    // A fragment: 16-bit A 16x32 layout -> lane holds row (lane&15),
    //   K = (e&7) + (e>=8 ? 16 : 0) + 8*half + 32*chunk
    const int w0 = chunk * 32 + half * 8;
    const float4* a4 = (const float4*)(arow + w0);
    float4 f0 = a4[0], f1 = a4[1], f2 = a4[4], f3 = a4[5];
    v16h a;
    a[0]=(_Float16)f0.x;  a[1]=(_Float16)f0.y;  a[2]=(_Float16)f0.z;  a[3]=(_Float16)f0.w;
    a[4]=(_Float16)f1.x;  a[5]=(_Float16)f1.y;  a[6]=(_Float16)f1.z;  a[7]=(_Float16)f1.w;
    a[8]=(_Float16)f2.x;  a[9]=(_Float16)f2.y;  a[10]=(_Float16)f2.z; a[11]=(_Float16)f2.w;
    a[12]=(_Float16)f3.x; a[13]=(_Float16)f3.y; a[14]=(_Float16)f3.z; a[15]=(_Float16)f3.w;

    // B fragment: lane holds column (lane&15) = rq row, K = 32*chunk + (e&7) + (e>=8 ? 16 : 0) + 8*half
    const int bw = chunk * 32 + half * 8;
    const v8h* bp = (const v8h*)&rqh[lrow * WC + bw];
    v8h b0 = bp[0], b1 = bp[2];
    v16h bf;
    #pragma unroll
    for (int e = 0; e < 8; ++e) { bf[e] = b0[e]; bf[e + 8] = b1[e]; }

    acc = __builtin_amdgcn_wmma_f32_16x16x32_f16(
        /*neg_a=*/false, a, /*neg_b=*/false, bf,
        /*c_mod=*/(short)0, acc, /*reuse_a=*/false, /*reuse_b=*/false);
  }

  // D layout: lane L: N=r=(L&15); VGPR v: M = v + 8*(L>>4)
  if (lrow < R_) {
    float* sp = scores + ((size_t)b * R_ + lrow) * M_ + rowbase + half * 8;
    *(float4*)(sp    ) = make_float4(acc[0], acc[1], acc[2], acc[3]);
    *(float4*)(sp + 4) = make_float4(acc[4], acc[5], acc[6], acc[7]);
  }
}

// K3b: recompute scores at the <=161 rewritten rows using vis_new
__global__ void k3b_fix_scores(const float* __restrict__ rqbuf,
                               const float* __restrict__ visnew,
                               const int*   __restrict__ rowmap,
                               const int*   __restrict__ read_positions,
                               float* __restrict__ scores) {
  const int b = blockIdx.x, t = threadIdx.x;
  for (int task = t; task < C_ * R_; task += 256) {
    int c = task >> 3, r = task & 7;
    int p = read_positions[b * C_ + c];
    if (rowmap[(size_t)b * M_ + p] == c) {          // c owns row p (last writer)
      const float* vr = visnew + ((size_t)b * C_ + c) * WC;
      const float* q  = rqbuf + b * (R_ * WC) + r * WC;
      float d = 0.0f;
      for (int w = 0; w < WC; ++w) d += vr[w] * q[w];
      scores[((size_t)b * R_ + r) * M_ + p] = d;
    }
  }
}

// ============================================================
// K4: top-16 of 16384 per (b,r); jax tie-break (lower index first)
// ============================================================
__device__ __forceinline__ bool kbetter(float v1, int i1, float v2, int i2) {
  return v1 > v2 || (v1 == v2 && i1 < i2);
}

__global__ void k4_topk(const float* __restrict__ scores, int* __restrict__ posbuf) {
  __shared__ float cand_v[256 * 16];
  __shared__ int   cand_i[256 * 16];
  __shared__ float rv[256];
  __shared__ int   ri[256], rt[256];
  __shared__ int   winner;

  const int br = blockIdx.x;
  const int b = br >> 3, r = br & 7;
  const float* s = scores + ((size_t)b * R_ + r) * M_;
  const int t = threadIdx.x;

  float lv[16]; int li[16];
  #pragma unroll
  for (int i = 0; i < 16; ++i) { lv[i] = -3.4e38f; li[i] = 0x7fffffff; }
  for (int j = 0; j < M_ / 256; ++j) {           // ascending index scan
    int m = t + j * 256;
    float v = s[m];
    if (v > lv[15]) {                            // strict: keeps earlier idx on ties
      int p = 15;
      while (p > 0 && v > lv[p - 1]) { lv[p] = lv[p - 1]; li[p] = li[p - 1]; --p; }
      lv[p] = v; li[p] = m;
    }
  }
  #pragma unroll
  for (int i = 0; i < 16; ++i) { cand_v[t * 16 + i] = lv[i]; cand_i[t * 16 + i] = li[i]; }
  __syncthreads();

  int head = 0;
  for (int round = 0; round < 16; ++round) {
    rv[t] = (head < 16) ? cand_v[t * 16 + head] : -3.4e38f;
    ri[t] = (head < 16) ? cand_i[t * 16 + head] : 0x7fffffff;
    rt[t] = t;
    __syncthreads();
    for (int off = 128; off > 0; off >>= 1) {
      if (t < off && kbetter(rv[t + off], ri[t + off], rv[t], ri[t])) {
        rv[t] = rv[t + off]; ri[t] = ri[t + off]; rt[t] = rt[t + off];
      }
      __syncthreads();
    }
    if (t == 0) { posbuf[b * C_ + r * 16 + round] = ri[0]; winner = rt[0]; }
    __syncthreads();
    if (t == winner) head++;
    __syncthreads();
  }
}

// ============================================================
// K5: gather vis (updated memory), cosine, softmax, weighted sum
// ============================================================
__global__ void k5_read(const float* __restrict__ memory,
                        const float* __restrict__ rqbuf,
                        const float* __restrict__ visnew,
                        const int*   __restrict__ rowmap,
                        const int*   __restrict__ posbuf,
                        const int*   __restrict__ least_used,
                        float* __restrict__ out) {
  __shared__ float s_rq[R_ * WC];     // 8KB
  __shared__ float s_cw[R_ * C_];     // cos -> softmax weights
  __shared__ float s_bn[R_];
  __shared__ int   s_pos[C_], s_src[C_];

  const int b = blockIdx.x, t = threadIdx.x;

  for (int i = t; i < R_ * WC; i += 256) s_rq[i] = rqbuf[b * (R_ * WC) + i];
  if (t < C_) {
    int p = posbuf[b * C_ + t];
    int lu0 = least_used[0];                       // reference clips by [0,0]
    p = p < 0 ? 0 : (p > lu0 ? lu0 : p);
    s_pos[t] = p;
    s_src[t] = rowmap[(size_t)b * M_ + p];
  }
  __syncthreads();

  if (t < R_) {
    float ss = 0.0f;
    for (int w = 0; w < WC; ++w) { float v = s_rq[t * WC + w]; ss += v * v; }
    s_bn[t] = sqrtf(ss) + 1e-6f;
  }
  __syncthreads();

  const int wid = t >> 5, lane = t & 31;
  for (int c = wid; c < C_; c += 8) {
    const float* vrow = (s_src[c] >= 0)
        ? visnew + ((size_t)b * C_ + s_src[c]) * WC
        : memory + ((size_t)b * M_ + s_pos[c]) * WC;
    float ss = 0.0f, dp[R_];
    #pragma unroll
    for (int r = 0; r < R_; ++r) dp[r] = 0.0f;
    #pragma unroll
    for (int j = 0; j < 8; ++j) {
      int w = lane * 8 + j;
      float v = vrow[w];
      ss += v * v;
      #pragma unroll
      for (int r = 0; r < R_; ++r) dp[r] += v * s_rq[r * WC + w];
    }
    for (int off = 16; off > 0; off >>= 1) {
      ss += __shfl_xor(ss, off, 32);
      #pragma unroll
      for (int r = 0; r < R_; ++r) dp[r] += __shfl_xor(dp[r], off, 32);
    }
    if (lane == 0) {
      float an = sqrtf(ss) + 1e-6f;
      #pragma unroll
      for (int r = 0; r < R_; ++r)
        s_cw[r * C_ + c] = dp[r] / (256.0f * s_bn[r] * an + 1e-6f);
    }
  }
  __syncthreads();

  if (t < R_) {                                   // softmax over c
    float mx = -3.4e38f;
    for (int c = 0; c < C_; ++c) mx = fmaxf(mx, s_cw[t * C_ + c]);
    float sum = 0.0f;
    for (int c = 0; c < C_; ++c) { float e = __expf(s_cw[t * C_ + c] - mx); s_cw[t * C_ + c] = e; sum += e; }
    float inv = 1.0f / sum;
    for (int c = 0; c < C_; ++c) s_cw[t * C_ + c] *= inv;
  }
  __syncthreads();

  {
    const int r = t >> 5;
    float acc[8];
    #pragma unroll
    for (int j = 0; j < 8; ++j) acc[j] = 0.0f;
    for (int c = 0; c < C_; ++c) {
      const float* vrow = (s_src[c] >= 0)
          ? visnew + ((size_t)b * C_ + s_src[c]) * WC
          : memory + ((size_t)b * M_ + s_pos[c]) * WC;
      float wgt = s_cw[r * C_ + c];
      #pragma unroll
      for (int j = 0; j < 8; ++j) acc[j] += wgt * vrow[lane * 8 + j];
    }
    #pragma unroll
    for (int j = 0; j < 8; ++j)
      out[(size_t)b * (R_ * WC) + r * WC + lane * 8 + j] = acc[j];
  }
}

// ============================================================
extern "C" void kernel_launch(void* const* d_in, const int* in_sizes, int n_in,
                              void* d_out, int out_size, void* d_ws, size_t ws_size,
                              hipStream_t stream) {
  (void)in_sizes; (void)n_in; (void)out_size; (void)ws_size;

  const float* x      = (const float*)d_in[0];
  const float* memory = (const float*)d_in[1];
  const float* vism   = (const float*)d_in[2];
  const float* link   = (const float*)d_in[3];
  const float* rev    = (const float*)d_in[4];
  const float* prec   = (const float*)d_in[5];
  const float* rws    = (const float*)d_in[6];
  const float* wws    = (const float*)d_in[7];
  const float* usage  = (const float*)d_in[8];
  const float* W_rq   = (const float*)d_in[9];
  const float* b_rq   = (const float*)d_in[10];
  const float* W_wv   = (const float*)d_in[11];
  const float* b_wv   = (const float*)d_in[12];
  const float* W_ig   = (const float*)d_in[13];
  const float* b_ig   = (const float*)d_in[14];
  const float* W_wg   = (const float*)d_in[15];
  const float* b_wg   = (const float*)d_in[16];
  const int*   rpos   = (const int*)d_in[17];
  const int*   lused  = (const int*)d_in[18];
  float* out = (float*)d_out;

  // workspace layout (floats then ints), ~24.5 MB total
  float* f = (float*)d_ws;
  float* rqbuf   = f;                       f += B_ * R_ * WC;     // 65536
  float* wvbuf   = f;                       f += B_ * WC;          // 8192
  float* gatebuf = f;                       f += B_ * C_;          // 5152
  float* wgbuf   = f;                       f += 32;               // padded
  float* scores  = f;                       f += (size_t)B_ * R_ * M_; // 4M
  float* visnew  = f;                       f += (size_t)B_ * C_ * WC; // 1.3M
  int* ip = (int*)f;
  int* rowmap = ip;                         ip += (size_t)B_ * M_;
  int* posbuf = ip;                         ip += B_ * C_;

  k1_gemms<<<(NOUT + 255) / 256, 256, 0, stream>>>(
      x, W_rq, b_rq, W_wv, b_wv, W_ig, b_ig, W_wg, b_wg,
      rqbuf, wvbuf, gatebuf, wgbuf);

  k2_state<<<B_, 256, 0, stream>>>(
      vism, link, rev, prec, rws, wws, usage, rpos, lused,
      gatebuf, wgbuf, wvbuf, visnew, rowmap, posbuf);

  k3_scores_wmma<<<dim3(M_ / 128, B_), 256, 0, stream>>>(memory, rqbuf, scores);

  k3b_fix_scores<<<B_, 256, 0, stream>>>(rqbuf, visnew, rowmap, rpos, scores);

  k4_topk<<<B_ * R_, 256, 0, stream>>>(scores, posbuf);

  k5_read<<<B_, 256, 0, stream>>>(memory, rqbuf, visnew, rowmap, posbuf, lused, out);
}
